// MogrifierLSTM_28647431864830
// MI455X (gfx1250) — compile-verified, matrix operand order checked by reading, not run on hardware
//
#include <hip/hip_runtime.h>
#include <hip/hip_bf16.h>
#include <stdint.h>

// ---------------------------------------------------------------------------
// Mogrifier LSTM for MI455X (gfx1250), wave32 + WMMA f32_16x16x32_f16.
// B=16, T=128, E=H=512, V=32000, S=5.
// ---------------------------------------------------------------------------

typedef __attribute__((ext_vector_type(16))) _Float16 v16h;
typedef __attribute__((ext_vector_type(8)))  _Float16 v8h;
typedef __attribute__((ext_vector_type(8)))  float    v8f;
typedef __attribute__((ext_vector_type(4)))  float    f4_t;

#define BB 16
#define TT 128
#define HH 512
#define VV 32000
#define SS 5

// ---- WMMA helper -----------------------------------------------------------
__device__ __forceinline__ v8f wmma16x16x32(v16h a, v16h b, v8f c) {
    return __builtin_amdgcn_wmma_f32_16x16x32_f16(
        /*neg_a=*/false, a, /*neg_b=*/false, b,
        /*c_mod=*/(short)0, c, /*reuse_a=*/false, /*reuse_b=*/false);
}

// Fragment layout (CDNA5 16-bit A/B matrix, wave32):
//   row/col = lane & 15 ; kb = (lane>>4)*8
//   halves 0..7  hold K = k0+kb+0..7    (16 contiguous bytes)
//   halves 8..15 hold K = k0+kb+16..23  (16 contiguous bytes, +32B offset)
// So one fragment = two b128 loads from row-major f16 [rows][ld] storage.

__device__ __forceinline__ v16h load_frag_f16(const _Float16* base, int ld,
                                              int row0, int k0) {
    int lane = threadIdx.x & 31;
    int hi = lane >> 4;
    const _Float16* p =
        base + (size_t)(row0 + (lane & 15)) * ld + k0 + hi * 8;
    v8h lo = *(const v8h*)(p);       // K = kb+0..7
    v8h up = *(const v8h*)(p + 16);  // K = kb+16..23
    v16h v;
#pragma unroll
    for (int i = 0; i < 8; ++i) {
        v[i] = lo[i];
        v[8 + i] = up[i];
    }
    return v;
}

// A fragment from f16 LDS mirror [16][512].
__device__ __forceinline__ v16h load_a_lds16(const _Float16* S16, int k0) {
    int lane = threadIdx.x & 31;
    int hi = lane >> 4;
    const _Float16* p = S16 + (lane & 15) * HH + k0 + hi * 8;
    v8h lo = *(const v8h*)(p);
    v8h up = *(const v8h*)(p + 16);
    v16h v;
#pragma unroll
    for (int i = 0; i < 8; ++i) {
        v[i] = lo[i];
        v[8 + i] = up[i];
    }
    return v;
}

// B fragment from row-major f32 weights (convert f32->f16 on load).
__device__ __forceinline__ v16h load_frag_f32(const float* base, int ld,
                                              int row0, int k0) {
    int lane = threadIdx.x & 31;
    int hi = lane >> 4;
    const float* p = base + (size_t)(row0 + (lane & 15)) * ld + k0 + hi * 8;
    f4_t a0 = *(const f4_t*)(p);
    f4_t a1 = *(const f4_t*)(p + 4);
    f4_t b0 = *(const f4_t*)(p + 16);
    f4_t b1 = *(const f4_t*)(p + 20);
    v16h v;
#pragma unroll
    for (int i = 0; i < 4; ++i) {
        v[i] = (_Float16)a0[i];
        v[4 + i] = (_Float16)a1[i];
        v[8 + i] = (_Float16)b0[i];
        v[12 + i] = (_Float16)b1[i];
    }
    return v;
}

__device__ __forceinline__ float fsigmoid(float z) {
    return __builtin_amdgcn_rcpf(1.0f + __builtin_amdgcn_exp2f(-1.44269504f * z));
}
__device__ __forceinline__ float ftanh(float z) {
    return 2.0f * fsigmoid(2.0f * z) - 1.0f;
}

// ---- Kernel 1: embedding gather into time-major f32 [T][B][E] --------------
__global__ void embed_kernel(const int* __restrict__ seq,
                             const float* __restrict__ emb,
                             float* __restrict__ xseq) {
    size_t idx = (size_t)blockIdx.x * blockDim.x + threadIdx.x;
    if (idx >= (size_t)TT * BB * HH) return;
    int e = idx & (HH - 1);
    int b = (idx >> 9) & (BB - 1);
    int t = idx >> 13;
    int tok = seq[b * TT + t];
    xseq[idx] = emb[(size_t)tok * HH + e];
}

// ---- Kernel 2: f32 -> f16 weight conversion --------------------------------
__global__ void cvt_f16_kernel(const float* __restrict__ src,
                               _Float16* __restrict__ dst, int n) {
    int i = blockIdx.x * blockDim.x + threadIdx.x;
    if (i < n) dst[i] = (_Float16)src[i];
}

// ---- Recurrent scan helpers (device) ---------------------------------------
// 32 waves, wave w owns N-tile columns [16w, 16w+16).

__device__ void mogrify_dev(float* xm, float* hm, _Float16* xm16,
                            _Float16* hm16, const _Float16* W,
                            const float* bias) {
    int wave = threadIdx.x >> 5;
    int lane = threadIdx.x & 31;
    int n0 = wave * 16;
    int n = n0 + (lane & 15);
    int hi = lane >> 4;
#pragma unroll 1
    for (int s = 0; s < SS; ++s) {
        const _Float16* src16 = (s & 1) ? xm16 : hm16;  // even: reads h
        float* dst = (s & 1) ? hm : xm;                 // even: updates x
        _Float16* dst16 = (s & 1) ? hm16 : xm16;
        const _Float16* Ws = W + (size_t)s * HH * HH;
        v8f acc = {};
        for (int k = 0; k < HH; k += 32) {
            v16h a = load_a_lds16(src16, k);
            v16h b = load_frag_f16(Ws, HH, n0, k);
            acc = wmma16x16x32(a, b, acc);
        }
        float bn = bias[s * HH + n];
#pragma unroll
        for (int r = 0; r < 8; ++r) {
            int m = r + hi * 8;
            float g = 2.0f * fsigmoid(acc[r] + bn);
            float nv = dst[m * HH + n] * g;
            dst[m * HH + n] = nv;
            dst16[m * HH + n] = (_Float16)nv;
        }
        __syncthreads();
    }
}

__device__ void lstm_dev(const _Float16* xm16, const _Float16* hm16, float* c,
                         float* hstate, const _Float16* Wih,
                         const _Float16* Whh, const float* bih,
                         const float* bhh) {
    int wave = threadIdx.x >> 5;
    int lane = threadIdx.x & 31;
    int n0 = wave * 16;
    int n = n0 + (lane & 15);
    int hi = lane >> 4;
    v8f acc[4] = {v8f{}, v8f{}, v8f{}, v8f{}};  // i, f, g, o share an N-range
    for (int k = 0; k < HH; k += 32) {
        v16h a = load_a_lds16(xm16, k);
#pragma unroll
        for (int q = 0; q < 4; ++q) {
            v16h b = load_frag_f16(Wih, HH, q * HH + n0, k);
            acc[q] = wmma16x16x32(a, b, acc[q]);
        }
    }
    for (int k = 0; k < HH; k += 32) {
        v16h a = load_a_lds16(hm16, k);
#pragma unroll
        for (int q = 0; q < 4; ++q) {
            v16h b = load_frag_f16(Whh, HH, q * HH + n0, k);
            acc[q] = wmma16x16x32(a, b, acc[q]);
        }
    }
    float bi = bih[n] + bhh[n];
    float bf = bih[HH + n] + bhh[HH + n];
    float bg = bih[2 * HH + n] + bhh[2 * HH + n];
    float bo = bih[3 * HH + n] + bhh[3 * HH + n];
#pragma unroll
    for (int r = 0; r < 8; ++r) {
        int m = r + hi * 8;
        float ii = fsigmoid(acc[0][r] + bi);
        float ff = fsigmoid(acc[1][r] + bf);
        float gg = ftanh(acc[2][r] + bg);
        float oo = fsigmoid(acc[3][r] + bo);
        float cn = ff * c[m * HH + n] + ii * gg;
        c[m * HH + n] = cn;
        hstate[m * HH + n] = oo * ftanh(cn);
    }
}

// ---- Kernel 3: persistent single-workgroup scan over T ---------------------
__global__ __launch_bounds__(1024) void scan_kernel(
    const float* __restrict__ xseq,              // [T][B][E] f32
    const _Float16* __restrict__ m1w, const float* __restrict__ m1b,
    const _Float16* __restrict__ wih1, const _Float16* __restrict__ whh1,
    const float* __restrict__ bih1, const float* __restrict__ bhh1,
    const _Float16* __restrict__ m2w, const float* __restrict__ m2b,
    const _Float16* __restrict__ wih2, const _Float16* __restrict__ whh2,
    const float* __restrict__ bih2, const float* __restrict__ bhh2,
    float* __restrict__ out_hidden,              // [B][T][H] f32
    _Float16* __restrict__ hidf16)               // [B*T][H] f16 (FC A-matrix)
{
    extern __shared__ float lds[];
    float* xm = lds;                 // [16][512] f32
    float* hm = xm + BB * HH;
    float* h1 = hm + BB * HH;
    float* c1 = h1 + BB * HH;
    float* h2 = c1 + BB * HH;
    float* c2 = h2 + BB * HH;                      // 192KB f32 state
    _Float16* xm16 = (_Float16*)(c2 + BB * HH);    // [16][512] f16 mirror
    _Float16* hm16 = xm16 + BB * HH;               // +32KB => 224KB total
    int tid = threadIdx.x;

    for (int i = tid; i < BB * HH; i += 1024) {
        h1[i] = 0.0f; c1[i] = 0.0f; h2[i] = 0.0f; c2[i] = 0.0f;
    }
    __syncthreads();

    for (int t = 0; t < TT; ++t) {
        // layer 1 inputs: xm = x_t, hm = copy of h1 (f32 + f16 mirror)
        for (int i = tid; i < BB * HH; i += 1024) {
            float xv = xseq[(size_t)t * BB * HH + i];
            float hv = h1[i];
            xm[i] = xv; xm16[i] = (_Float16)xv;
            hm[i] = hv; hm16[i] = (_Float16)hv;
        }
        __syncthreads();
        mogrify_dev(xm, hm, xm16, hm16, m1w, m1b);
        lstm_dev(xm16, hm16, c1, h1, wih1, whh1, bih1, bhh1);
        __syncthreads();
        // layer 2 inputs: xm = copy of h1, hm = copy of h2
        for (int i = tid; i < BB * HH; i += 1024) {
            float xv = h1[i];
            float hv = h2[i];
            xm[i] = xv; xm16[i] = (_Float16)xv;
            hm[i] = hv; hm16[i] = (_Float16)hv;
        }
        __syncthreads();
        mogrify_dev(xm, hm, xm16, hm16, m2w, m2b);
        lstm_dev(xm16, hm16, c2, h2, wih2, whh2, bih2, bhh2);
        __syncthreads();
        // emit h2 for this timestep
        for (int i = tid; i < BB * HH; i += 1024) {
            int b = i >> 9;
            int hh = i & (HH - 1);
            float v = h2[i];
            size_t m = (size_t)b * TT + t;
            out_hidden[m * HH + hh] = v;
            hidf16[m * HH + hh] = (_Float16)v;
        }
        __syncthreads();
    }
}

// ---- Kernel 4: FC projection GEMM [2048,512] x [32000,512]^T + bias --------
// Block: 256 threads = 8 waves; each wave computes one 16x32 strip
// (two 16x16 tiles, A fragment reused). Block covers 16 x 256 output.
__global__ __launch_bounds__(256) void fc_kernel_f16(
    const _Float16* __restrict__ A,   // [2048][512] f16
    const _Float16* __restrict__ W,   // [32000][512] f16
    const float* __restrict__ bfc,    // [32000]
    float* __restrict__ out)          // [2048][32000]
{
    int mt = blockIdx.x;                       // 0..127
    int nb = blockIdx.y;                       // 0..124
    int wave = threadIdx.x >> 5;               // 8 waves
    int lane = threadIdx.x & 31;
    int m0 = mt * 16;
    int n0 = nb * 256 + wave * 32;
    v8f acc0 = {}, acc1 = {};
    for (int k = 0; k < HH; k += 32) {
        v16h a = load_frag_f16(A, HH, m0, k);
        v16h b0 = load_frag_f16(W, HH, n0, k);
        v16h b1 = load_frag_f16(W, HH, n0 + 16, k);
        acc0 = wmma16x16x32(a, b0, acc0);
        acc1 = wmma16x16x32(a, b1, acc1);
    }
    int nc = lane & 15;
    int hi = lane >> 4;
    float bn0 = bfc[n0 + nc];
    float bn1 = bfc[n0 + 16 + nc];
#pragma unroll
    for (int r = 0; r < 8; ++r) {
        int m = m0 + r + hi * 8;
        out[(size_t)m * VV + n0 + nc] = acc0[r] + bn0;
        out[(size_t)m * VV + n0 + 16 + nc] = acc1[r] + bn1;
    }
}

__global__ __launch_bounds__(256) void fc_kernel_f32(
    const _Float16* __restrict__ A,   // [2048][512] f16
    const float* __restrict__ W,      // [32000][512] f32
    const float* __restrict__ bfc,    // [32000]
    float* __restrict__ out)          // [2048][32000]
{
    int mt = blockIdx.x;
    int nb = blockIdx.y;
    int wave = threadIdx.x >> 5;
    int lane = threadIdx.x & 31;
    int m0 = mt * 16;
    int n0 = nb * 256 + wave * 32;
    v8f acc0 = {}, acc1 = {};
    for (int k = 0; k < HH; k += 32) {
        v16h a = load_frag_f16(A, HH, m0, k);
        v16h b0 = load_frag_f32(W, HH, n0, k);
        v16h b1 = load_frag_f32(W, HH, n0 + 16, k);
        acc0 = wmma16x16x32(a, b0, acc0);
        acc1 = wmma16x16x32(a, b1, acc1);
    }
    int nc = lane & 15;
    int hi = lane >> 4;
    float bn0 = bfc[n0 + nc];
    float bn1 = bfc[n0 + 16 + nc];
#pragma unroll
    for (int r = 0; r < 8; ++r) {
        int m = m0 + r + hi * 8;
        out[(size_t)m * VV + n0 + nc] = acc0[r] + bn0;
        out[(size_t)m * VV + n0 + 16 + nc] = acc1[r] + bn1;
    }
}

// ---------------------------------------------------------------------------
extern "C" void kernel_launch(void* const* d_in, const int* in_sizes, int n_in,
                              void* d_out, int out_size, void* d_ws,
                              size_t ws_size, hipStream_t stream) {
    const int*   seq    = (const int*)d_in[0];
    const float* emb    = (const float*)d_in[1];
    const float* mog1_W = (const float*)d_in[2];
    const float* mog1_b = (const float*)d_in[3];
    const float* Wih1   = (const float*)d_in[4];
    const float* Whh1   = (const float*)d_in[5];
    const float* bih1   = (const float*)d_in[6];
    const float* bhh1   = (const float*)d_in[7];
    const float* mog2_W = (const float*)d_in[8];
    const float* mog2_b = (const float*)d_in[9];
    const float* Wih2   = (const float*)d_in[10];
    const float* Whh2   = (const float*)d_in[11];
    const float* bih2   = (const float*)d_in[12];
    const float* bhh2   = (const float*)d_in[13];
    const float* Wfc    = (const float*)d_in[14];
    const float* bfc    = (const float*)d_in[15];

    float* outputs    = (float*)d_out;                          // [B][T][V]
    float* out_hidden = outputs + (size_t)BB * TT * VV;         // [B][T][H]

    // workspace layout (byte offsets, all 16B aligned)
    char* ws = (char*)d_ws;
    float*     xseq   = (float*)(ws + 0);                       // 4 MB
    _Float16*  m1w16  = (_Float16*)(ws + 4194304);              // 2.62 MB
    _Float16*  m2w16  = (_Float16*)(ws + 6815744);              // 2.62 MB
    _Float16*  wih1_h = (_Float16*)(ws + 9437184);              // 2 MB
    _Float16*  whh1_h = (_Float16*)(ws + 11534336);             // 2 MB
    _Float16*  wih2_h = (_Float16*)(ws + 13631488);             // 2 MB
    _Float16*  whh2_h = (_Float16*)(ws + 15728640);             // 2 MB
    _Float16*  hidf16 = (_Float16*)(ws + 17825792);             // 2 MB
    _Float16*  wfc16  = (_Float16*)(ws + 19922944);             // 32.77 MB opt
    const size_t WS_NEED_WFC16 = 19922944ull + (size_t)VV * HH * 2ull;
    const bool use_wfc16 = (ws_size >= WS_NEED_WFC16);

    // 1) embedding gather (time-major f32)
    {
        size_t total = (size_t)TT * BB * HH;
        embed_kernel<<<(unsigned)((total + 255) / 256), 256, 0, stream>>>(
            seq, emb, xseq);
    }
    // 2) weight conversions f32 -> f16
    {
        const int nM = SS * HH * HH;     // 1,310,720
        const int nW = 4 * HH * HH;      // 1,048,576
        cvt_f16_kernel<<<(nM + 255) / 256, 256, 0, stream>>>(mog1_W, m1w16, nM);
        cvt_f16_kernel<<<(nM + 255) / 256, 256, 0, stream>>>(mog2_W, m2w16, nM);
        cvt_f16_kernel<<<(nW + 255) / 256, 256, 0, stream>>>(Wih1, wih1_h, nW);
        cvt_f16_kernel<<<(nW + 255) / 256, 256, 0, stream>>>(Whh1, whh1_h, nW);
        cvt_f16_kernel<<<(nW + 255) / 256, 256, 0, stream>>>(Wih2, wih2_h, nW);
        cvt_f16_kernel<<<(nW + 255) / 256, 256, 0, stream>>>(Whh2, whh2_h, nW);
        if (use_wfc16) {
            const int nF = VV * HH;      // 16,384,000
            cvt_f16_kernel<<<(nF + 255) / 256, 256, 0, stream>>>(Wfc, wfc16, nF);
        }
    }
    // 3) persistent sequential scan: 1 workgroup, 32 waves, 224KB LDS
    {
        size_t lds_bytes = (size_t)(6 * BB * HH) * sizeof(float) +
                           (size_t)(2 * BB * HH) * sizeof(_Float16);
        scan_kernel<<<1, 1024, lds_bytes, stream>>>(
            xseq, m1w16, mog1_b, wih1_h, whh1_h, bih1, bhh1, m2w16, mog2_b,
            wih2_h, whh2_h, bih2, bhh2, out_hidden, hidf16);
    }
    // 4) output projection GEMM
    {
        dim3 grid((BB * TT) / 16, VV / 256);   // 128 x 125
        if (use_wfc16) {
            fc_kernel_f16<<<grid, 256, 0, stream>>>(hidf16, wfc16, bfc, outputs);
        } else {
            fc_kernel_f32<<<grid, 256, 0, stream>>>(hidf16, Wfc, bfc, outputs);
        }
    }
}
